// Decoder_43387759624346
// MI455X (gfx1250) — matrix operand order repaired and checked by reading, not compiled, not run on hardware
//
#include <hip/hip_runtime.h>
#include <hip/hip_fp16.h>

typedef __attribute__((ext_vector_type(16))) _Float16     v16h;
typedef __attribute__((ext_vector_type(8)))  float        v8f;
typedef __attribute__((ext_vector_type(4)))  unsigned int v4u;

#define DEVINL __device__ __forceinline__

namespace {

constexpr int B_ = 2048, T_ = 64, C_ = 9, H_ = 512, Z_ = 256;
constexpr long long BTC = (long long)B_ * T_ * C_;
constexpr float LEAK = 0.2f;

// LDS strides (in halves) padded so 16-lane row gathers avoid bank collisions
constexpr int HS = 520;   // for H=512 wide activation rows
constexpr int XS = 72;    // for 64-wide padded GRU1 input
constexpr int ZS = 264;   // for Z=256 wide latent rows (prologue staging)

// dynamic-LDS byte offsets (all 16B aligned)
constexpr int HBUF_BYTES = 16 * HS * 2;
constexpr int OFF_H1_0 = 0;
constexpr int OFF_H1_1 = OFF_H1_0 + HBUF_BYTES;
constexpr int OFF_H2_0 = OFF_H1_1 + HBUF_BYTES;
constexpr int OFF_H2_1 = OFF_H2_0 + HBUF_BYTES;
constexpr int OFF_O    = OFF_H2_1 + HBUF_BYTES;
constexpr int OFF_X    = OFF_O + 16 * HS * 2;
constexpr int OFF_HIT  = OFF_X + 16 * XS * 2;
constexpr int OFF_VELB = OFF_HIT + 16 * 16 * 4;
constexpr int OFF_MTB  = OFF_VELB + 16 * 16 * 4;
constexpr int SMEM_BYTES = OFF_MTB + 16 * 16 * 4;   // ~88.5 KB

// packed-weight offsets in d_ws (units: _Float16). Tile = 32 lanes x 16 halves.
constexpr size_t OFF_G1WIH = 0;                          // 96 nt x 2 kt
constexpr size_t OFF_G1WHH = OFF_G1WIH + (size_t)96*2*512;
constexpr size_t OFF_G2WIH = OFF_G1WHH + (size_t)96*16*512;
constexpr size_t OFF_G2WHH = OFF_G2WIH + (size_t)96*16*512;
constexpr size_t OFF_OUTW  = OFF_G2WHH + (size_t)96*16*512;
constexpr size_t OFF_INW   = OFF_OUTW + (size_t)32*16*512;
constexpr size_t OFF_NOTE  = OFF_INW  + (size_t)32*8*512;
constexpr size_t OFF_VELW  = OFF_NOTE + (size_t)16*512;
constexpr size_t OFF_MTW   = OFF_VELW + (size_t)16*512;
// total = 2,875,392 halves = ~5.75 MB of d_ws

union Frag { v16h v; v4u q[2]; };

// A-fragment (16x32 f16): lane<16 -> M=lane, K = k0..k0+7 and k0+16..k0+23
// with k0 = kt*32 (+8 for upper half-wave). Two ds_load_b128 per lane.
DEVINL v16h load_a(const _Float16* base, int stride, int kt, int lane) {
  int m  = lane & 15;
  int k0 = kt * 32 + ((lane >> 4) << 3);
  const _Float16* p = base + m * stride + k0;
  Frag f;
  f.q[0] = *(const v4u*)(p);
  f.q[1] = *(const v4u*)(p + 16);
  return f.v;
}

// B-fragment (32x16 f16) from pre-packed global: 32 bytes/lane, contiguous.
DEVINL v16h load_b(const _Float16* packed, size_t tile, int lane) {
  const v4u* p = (const v4u*)(packed + (tile * 32 + (size_t)lane) * 16);
  Frag f;
  f.q[0] = p[0];
  f.q[1] = p[1];
  return f.v;
}

DEVINL v8f wmma(v16h a, v16h b, v8f c) {
  return __builtin_amdgcn_wmma_f32_16x16x32_f16(
      /*neg_a=*/false, a, /*neg_b=*/false, b,
      /*c_mod=*/(short)0, c, /*reuse_a=*/false, /*reuse_b=*/false);
}

// Hardware transcendental tanh on gfx1250 (V_TANH_F32); graceful fallback.
DEVINL float tanh_hw(float x) {
#if __has_builtin(__builtin_amdgcn_tanhf)
  return __builtin_amdgcn_tanhf(x);
#else
  return tanhf(x);
#endif
}
DEVINL float sig(float x) { return __builtin_fmaf(0.5f, tanh_hw(0.5f * x), 0.5f); }

// Pack fp32 weight W[N,K] (row = output unit) into WMMA-B fragment order:
// dst[((nt*Ktiles+kt)*32 + lane)*16 + e] = f16(W[nt*16+(lane&15), kt*32+((lane>>4)<<3)+(e&7)+((e>>3)<<4)])
__global__ void pack_w_kernel(const float* __restrict__ src, _Float16* __restrict__ dst,
                              int N, int K, int Ktiles, int total) {
  int idx = blockIdx.x * blockDim.x + threadIdx.x;
  if (idx >= total) return;
  int e    = idx & 15;
  int lane = (idx >> 4) & 31;
  int tile = idx >> 9;
  int kt   = tile % Ktiles;
  int nt   = tile / Ktiles;
  int n = nt * 16 + (lane & 15);
  int k = kt * 32 + ((lane >> 4) << 3) + (e & 7) + ((e >> 3) << 4);
  float v = (n < N && k < K) ? src[(size_t)n * K + k] : 0.f;
  dst[idx] = (_Float16)v;
}

__global__ __launch_bounds__(256) void decoder_persistent_kernel(
    const float* __restrict__ z, const float* __restrict__ skel, const float* __restrict__ rnd,
    const float* __restrict__ in_b,
    const float* __restrict__ g1_bih, const float* __restrict__ g1_bhh,
    const float* __restrict__ g2_bih, const float* __restrict__ g2_bhh,
    const float* __restrict__ out_b,
    const float* __restrict__ note_b, const float* __restrict__ vel_b, const float* __restrict__ mt_b,
    const _Float16* __restrict__ ws, float* __restrict__ out) {
  extern __shared__ unsigned char smraw[];
  _Float16* obuf = (_Float16*)(smraw + OFF_O);
  _Float16* xbuf = (_Float16*)(smraw + OFF_X);
  float* hitb = (float*)(smraw + OFF_HIT);
  float* velb = (float*)(smraw + OFF_VELB);
  float* mtb  = (float*)(smraw + OFF_MTB);

  const int tid  = threadIdx.x;
  const int lane = tid & 31;
  const int wave = tid >> 5;
  const int b0   = blockIdx.x * 16;

  const _Float16* wG1WIH = ws + OFF_G1WIH;
  const _Float16* wG1WHH = ws + OFF_G1WHH;
  const _Float16* wG2WIH = ws + OFF_G2WIH;
  const _Float16* wG2WHH = ws + OFF_G2WHH;
  const _Float16* wOUTW  = ws + OFF_OUTW;
  const _Float16* wINW   = ws + OFF_INW;

  // ---------------- prologue: h1 = tanh(z @ in_w.T + in_b) ----------------
  _Float16* zst = obuf;  // reuse o-buffer region for z staging (stride ZS)
  for (int idx = tid; idx < 16 * Z_; idx += 256) {
    int m = idx >> 8, k = idx & 255;
    zst[m * ZS + k] = (_Float16)z[(size_t)(b0 + m) * Z_ + k];
  }
  // x0 = [zeros(27), skel[:,0,:], zeros pad to 64]
  for (int idx = tid; idx < 16 * 64; idx += 256) {
    int m = idx >> 6, k = idx & 63;
    float v = (k >= 27 && k < 36) ? skel[(size_t)(b0 + m) * (T_ * C_) + (k - 27)] : 0.f;
    xbuf[m * XS + k] = (_Float16)v;
  }
  __syncthreads();
  {
    _Float16* h1init = (_Float16*)(smraw + OFF_H1_0);
    for (int nt = wave; nt < 32; nt += 8) {
      v8f acc = {};
      for (int kt = 0; kt < 8; ++kt)
        acc = wmma(load_a(zst, ZS, kt, lane), load_b(wINW, (size_t)nt * 8 + kt, lane), acc);
      int n = nt * 16 + (lane & 15);
      float bb = in_b[n];
      int mr = (lane >> 4) << 3;
      for (int i = 0; i < 8; ++i)
        h1init[(mr + i) * HS + n] = (_Float16)tanh_hw(acc[i] + bb);
    }
  }
  __syncthreads();

  // ---------------- T-step recurrence ----------------
  for (int t = 0; t < T_; ++t) {
    const int cur = t & 1;
    _Float16* h1cur = (_Float16*)(smraw + (cur ? OFF_H1_1 : OFF_H1_0));
    _Float16* h1nxt = (_Float16*)(smraw + (cur ? OFF_H1_0 : OFF_H1_1));
    _Float16* h2cur = (_Float16*)(smraw + (cur ? OFF_H2_1 : OFF_H2_0));
    _Float16* h2nxt = (_Float16*)(smraw + (cur ? OFF_H2_0 : OFF_H2_1));

    // ---- GRU1: x[16x64] @ g1_wih.T + h1[16x512] @ g1_whh.T ----
    for (int nt = wave; nt < 32; nt += 8) {
      v8f ar = {}, au = {}, ai = {}, ah = {};
      for (int kt = 0; kt < 16; ++kt) {
        v16h a = load_a(h1cur, HS, kt, lane);
        ar = wmma(a, load_b(wG1WHH, (size_t)(nt)      * 16 + kt, lane), ar);
        au = wmma(a, load_b(wG1WHH, (size_t)(nt + 32) * 16 + kt, lane), au);
        ah = wmma(a, load_b(wG1WHH, (size_t)(nt + 64) * 16 + kt, lane), ah);
      }
      for (int kt = 0; kt < 2; ++kt) {
        v16h a = load_a(xbuf, XS, kt, lane);
        ar = wmma(a, load_b(wG1WIH, (size_t)(nt)      * 2 + kt, lane), ar);
        au = wmma(a, load_b(wG1WIH, (size_t)(nt + 32) * 2 + kt, lane), au);
        ai = wmma(a, load_b(wG1WIH, (size_t)(nt + 64) * 2 + kt, lane), ai);
      }
      int n = nt * 16 + (lane & 15);
      float br = g1_bih[n] + g1_bhh[n];
      float bu = g1_bih[H_ + n] + g1_bhh[H_ + n];
      float bi = g1_bih[2 * H_ + n];
      float bh = g1_bhh[2 * H_ + n];
      int mr = (lane >> 4) << 3;
      for (int i = 0; i < 8; ++i) {
        float r  = sig(ar[i] + br);
        float u  = sig(au[i] + bu);
        float nn = tanh_hw(ai[i] + bi + r * (ah[i] + bh));
        float ho = (float)h1cur[(mr + i) * HS + n];
        h1nxt[(mr + i) * HS + n] = (_Float16)((1.f - u) * nn + u * ho);
      }
    }
    __syncthreads();

    if (t == 0) {  // h2 initialized from h1n at step 0
      const unsigned int* s = (const unsigned int*)h1nxt;
      unsigned int* d = (unsigned int*)h2cur;
      for (int idx = tid; idx < 16 * HS / 2; idx += 256) d[idx] = s[idx];
      __syncthreads();
    }

    // ---- GRU2: h1n @ g2_wih.T + h2 @ g2_whh.T ----
    for (int nt = wave; nt < 32; nt += 8) {
      v8f ar = {}, au = {}, ai = {}, ah = {};
      for (int kt = 0; kt < 16; ++kt) {
        v16h a1 = load_a(h1nxt, HS, kt, lane);
        v16h a2 = load_a(h2cur, HS, kt, lane);
        ar = wmma(a1, load_b(wG2WIH, (size_t)(nt)      * 16 + kt, lane), ar);
        ar = wmma(a2, load_b(wG2WHH, (size_t)(nt)      * 16 + kt, lane), ar);
        au = wmma(a1, load_b(wG2WIH, (size_t)(nt + 32) * 16 + kt, lane), au);
        au = wmma(a2, load_b(wG2WHH, (size_t)(nt + 32) * 16 + kt, lane), au);
        ai = wmma(a1, load_b(wG2WIH, (size_t)(nt + 64) * 16 + kt, lane), ai);
        ah = wmma(a2, load_b(wG2WHH, (size_t)(nt + 64) * 16 + kt, lane), ah);
      }
      int n = nt * 16 + (lane & 15);
      float br = g2_bih[n] + g2_bhh[n];
      float bu = g2_bih[H_ + n] + g2_bhh[H_ + n];
      float bi = g2_bih[2 * H_ + n];
      float bh = g2_bhh[2 * H_ + n];
      int mr = (lane >> 4) << 3;
      for (int i = 0; i < 8; ++i) {
        float r  = sig(ar[i] + br);
        float u  = sig(au[i] + bu);
        float nn = tanh_hw(ai[i] + bi + r * (ah[i] + bh));
        float ho = (float)h2cur[(mr + i) * HS + n];
        h2nxt[(mr + i) * HS + n] = (_Float16)((1.f - u) * nn + u * ho);
      }
    }
    __syncthreads();

    // ---- out: o = LeakyReLU(h2n @ out_w.T + out_b) ----
    for (int nt = wave; nt < 32; nt += 8) {
      v8f acc = {};
      for (int kt = 0; kt < 16; ++kt)
        acc = wmma(load_a(h2nxt, HS, kt, lane), load_b(wOUTW, (size_t)nt * 16 + kt, lane), acc);
      int n = nt * 16 + (lane & 15);
      float bb = out_b[n];
      int mr = (lane >> 4) << 3;
      for (int i = 0; i < 8; ++i) {
        float v = acc[i] + bb;
        v = v > 0.f ? v : LEAK * v;
        obuf[(mr + i) * HS + n] = (_Float16)v;
      }
    }
    __syncthreads();

    // ---- heads: note / vel / mt (one wave each), write out + sample prev ----
    if (wave < 3) {
      const _Float16* wp = (wave == 0) ? (ws + OFF_NOTE)
                         : (wave == 1) ? (ws + OFF_VELW)
                                       : (ws + OFF_MTW);
      v8f acc = {};
      for (int kt = 0; kt < 16; ++kt)
        acc = wmma(load_a(obuf, HS, kt, lane), load_b(wp, (size_t)kt, lane), acc);
      int c = lane & 15;
      int mr = (lane >> 4) << 3;
      if (c < C_) {
        float bb = (wave == 0) ? note_b[c] : (wave == 1) ? vel_b[c] : mt_b[c];
        for (int i = 0; i < 8; ++i) {
          int m = mr + i;
          size_t b = (size_t)(b0 + m);
          size_t oidx = b * (size_t)(T_ * C_) + (size_t)t * C_ + c;
          float v = acc[i] + bb;
          if (wave == 0) {
            out[oidx] = v;  // raw note logits
            float s  = sig(v);
            float rv = rnd[((size_t)t * B_ + b) * C_ + c];
            hitb[m * 16 + c] = (s - rv > 0.f) ? 1.f : 0.f;
          } else if (wave == 1) {
            float s = sig(v);
            out[(size_t)BTC + oidx] = s;
            velb[m * 16 + c] = s;
          } else {
            float s = tanh_hw(v);
            out[2 * (size_t)BTC + oidx] = s;
            mtb[m * 16 + c] = s;
          }
        }
      }
    }
    __syncthreads();

    // ---- rebuild x = [hit, hit*vel, hit*mt, skel[:,t+1,:]] (padded to 64) ----
    if (t + 1 < T_) {
      for (int idx = tid; idx < 16 * 36; idx += 256) {
        int m = idx / 36, k = idx - m * 36;
        float v;
        if      (k < 9)  v = hitb[m * 16 + k];
        else if (k < 18) v = hitb[m * 16 + (k - 9)]  * velb[m * 16 + (k - 9)];
        else if (k < 27) v = hitb[m * 16 + (k - 18)] * mtb[m * 16 + (k - 18)];
        else             v = skel[(size_t)(b0 + m) * (T_ * C_) + (size_t)(t + 1) * C_ + (k - 27)];
        xbuf[m * XS + k] = (_Float16)v;
      }
    }
    __syncthreads();
  }
}

}  // anonymous namespace

extern "C" void kernel_launch(void* const* d_in, const int* in_sizes, int n_in,
                              void* d_out, int out_size, void* d_ws, size_t ws_size,
                              hipStream_t stream) {
  (void)in_sizes; (void)n_in; (void)out_size; (void)ws_size;
  const float* z      = (const float*)d_in[0];
  const float* skel   = (const float*)d_in[1];
  const float* rnd    = (const float*)d_in[2];
  const float* in_w   = (const float*)d_in[3];
  const float* in_b   = (const float*)d_in[4];
  const float* g1_wih = (const float*)d_in[5];
  const float* g1_whh = (const float*)d_in[6];
  const float* g1_bih = (const float*)d_in[7];
  const float* g1_bhh = (const float*)d_in[8];
  const float* g2_wih = (const float*)d_in[9];
  const float* g2_whh = (const float*)d_in[10];
  const float* g2_bih = (const float*)d_in[11];
  const float* g2_bhh = (const float*)d_in[12];
  const float* out_w  = (const float*)d_in[13];
  const float* out_b  = (const float*)d_in[14];
  const float* note_w = (const float*)d_in[15];
  const float* note_b = (const float*)d_in[16];
  const float* vel_w  = (const float*)d_in[17];
  const float* vel_b  = (const float*)d_in[18];
  const float* mt_w   = (const float*)d_in[19];
  const float* mt_b   = (const float*)d_in[20];
  _Float16* ws = (_Float16*)d_ws;

  auto pack = [&](const float* src, size_t off, int N, int K, int Ntiles, int Ktiles) {
    int total = Ntiles * Ktiles * 512;
    pack_w_kernel<<<(total + 255) / 256, 256, 0, stream>>>(src, ws + off, N, K, Ktiles, total);
  };
  pack(g1_wih, OFF_G1WIH, 1536,  36, 96,  2);
  pack(g1_whh, OFF_G1WHH, 1536, 512, 96, 16);
  pack(g2_wih, OFF_G2WIH, 1536, 512, 96, 16);
  pack(g2_whh, OFF_G2WHH, 1536, 512, 96, 16);
  pack(out_w,  OFF_OUTW,   512, 512, 32, 16);
  pack(in_w,   OFF_INW,    512, 256, 32,  8);
  pack(note_w, OFF_NOTE,     9, 512,  1, 16);
  pack(vel_w,  OFF_VELW,     9, 512,  1, 16);
  pack(mt_w,   OFF_MTW,      9, 512,  1, 16);

  (void)hipFuncSetAttribute(reinterpret_cast<const void*>(decoder_persistent_kernel),
                            hipFuncAttributeMaxDynamicSharedMemorySize, SMEM_BYTES);
  decoder_persistent_kernel<<<B_ / 16, 256, SMEM_BYTES, stream>>>(
      z, skel, rnd, in_b, g1_bih, g1_bhh, g2_bih, g2_bhh, out_b,
      note_b, vel_b, mt_b, ws, (float*)d_out);
}
